// Seq2SeqEnc_19078244729064
// MI455X (gfx1250) — matrix operand order repaired and checked by reading, not compiled, not run on hardware
//
#include <hip/hip_runtime.h>
#include <hip/hip_bf16.h>

typedef __attribute__((ext_vector_type(16))) __bf16 v16bf;
typedef __attribute__((ext_vector_type(8)))  __bf16 v8bf;
typedef __attribute__((ext_vector_type(8)))  float  v8f;

#define B_  4
#define T_  12
#define H_  60
#define W_  80
#define HW_ (H_*W_)
#define F_  64
#define G_  256   /* 4*F gates */

__device__ __forceinline__ __bf16 f2bf(float f){
  unsigned u = __builtin_bit_cast(unsigned, f);
  u += 0x7fffu + ((u >> 16) & 1u);                 // round-to-nearest-even
  unsigned short h = (unsigned short)(u >> 16);
  return __builtin_bit_cast(__bf16, h);
}

__device__ __forceinline__ v16bf cat8(v8bf lo, v8bf hi){
  return __builtin_shufflevector(lo, hi, 0,1,2,3,4,5,6,7,8,9,10,11,12,13,14,15);
}

// ---------------------------------------------------------------------------
// Pack fp32 conv weights [256][Cin][3][3] into WMMA A-fragment order (bf16):
// linear index = ((mtile*nchunks + chunk)*32 + lane)*16 + j
// K-order: k = tap*Cin + c   (tap = ky*3+kx, channel-minor), zero for tap>=9.
// A-fragment layout (16-bit A 16x32, ISA 7.12.2):
//   lane<16 : j<8 -> K=j ; j>=8 -> K=j+8      lane>=16: +8 on K
// ---------------------------------------------------------------------------
__global__ void pack_weights(const float* __restrict__ Wsrc, __bf16* __restrict__ dst,
                             int nchunks, int cmask, int shift){
  int total = 16*nchunks*512;
  int i = blockIdx.x*blockDim.x + threadIdx.x;
  if (i >= total) return;
  int j    = i & 15;
  int lane = (i >> 4) & 31;
  int mc   = i >> 9;                 // mtile*nchunks + chunk
  int chunk = mc % nchunks;
  int mtile = mc / nchunks;
  int K = chunk*32 + (j & 7) + ((j >= 8) ? 16 : 0) + ((lane >= 16) ? 8 : 0);
  int cout = mtile*16 + (lane & 15);
  int tap = K >> shift;
  int c   = K & cmask;
  float v = (tap < 9) ? Wsrc[(long)(cout*(cmask+1) + c)*9 + tap] : 0.0f;
  dst[i] = f2bf(v);
}

// ---------------------------------------------------------------------------
// Implicit-GEMM 3x3 conv via v_wmma_f32_16x16x32_bf16.
// One workgroup (256 thr = 8 waves) computes all 256 out channels for a
// 2-row x 16-col pixel tile. Wave w handles M-tiles {2w, 2w+1} for both rows:
// 4 accumulators (2 M x 2 rows), 4 WMMAs per K-chunk, weights loaded once
// per (M-tile, chunk). Both weight streams use immediate-offset addressing.
// padclamp=1 -> symmetric(edge-clamp) pad; 0 -> zero pad (SAME).
// out[idx] = (addflag ? addsrc[idx] : 0) + conv ; addsrc may alias out.
// ---------------------------------------------------------------------------
__global__ void conv3x3_wmma(const float* __restrict__ src, long srcBatchStride,
                             const __bf16* __restrict__ wpack,
                             const float* __restrict__ addsrc, int addflag,
                             float* __restrict__ out,
                             int Ktot, int cmask, int shift, int padclamp){
  extern __shared__ __bf16 bls[];           // [32 px][Ktot+8] halfwords
  const int tid = threadIdx.x;
  const int b  = blockIdx.z;
  const int y0 = blockIdx.y * 2;
  const int x0 = blockIdx.x * 16;
  const int srow = Ktot + 8;                // hw stride; %16B-groups breaks bank conflicts
  const float* sb = src + (long)b * srcBatchStride;

  // ---- stage im2col tile (2 output rows) into LDS, zero-filled padded K ----
  for (int i = tid; i < (Ktot << 5); i += 256){
    int n = i & 31;                         // n>>4 = row, n&15 = col
    int k = i >> 5;
    int tap = k >> shift;
    float v = 0.0f;
    if (tap < 9){
      int c  = k & cmask;
      int ky = tap / 3, kx = tap - ky*3;
      int yy = y0 + (n >> 4) + ky - 1;
      int xx = x0 + (n & 15) + kx - 1;
      if (padclamp){
        yy = yy < 0 ? 0 : (yy > H_-1 ? H_-1 : yy);
        xx = xx < 0 ? 0 : (xx > W_-1 ? W_-1 : xx);
        v = sb[((long)c*H_ + yy)*W_ + xx];
      } else if (yy >= 0 && yy < H_ && xx >= 0 && xx < W_){
        v = sb[((long)c*H_ + yy)*W_ + xx];
      }
    }
    bls[n*srow + k] = f2bf(v);
  }
  __syncthreads();

  // ---- WMMA main loop ----
  const int lane = tid & 31;
  const int wave = tid >> 5;
  const int m0 = wave*2;
  const int nchunks = Ktot >> 5;
  const int nn   = lane & 15;
  const int hi16 = lane >> 4;
  // per-lane weight bases: chunk stride = 64 v8bf (1024B) -> immediate offsets
  const v8bf* wpl0 = (const v8bf*)wpack + (long)m0*nchunks*64 + lane*2;
  const v8bf* wpl1 = wpl0 + (long)nchunks*64;
  // B layout: lanes 0-15 hold K=0..7,16..23; lanes 16-31 hold K=8..15,24..31
  const __bf16* bb0 = bls + nn*srow + (hi16 << 3);          // row 0
  const __bf16* bb1 = bb0 + 16*srow;                        // row 1

  v8f acc00 = {}, acc01 = {}, acc10 = {}, acc11 = {};
  for (int ch = 0; ch < nchunks; ++ch){
    v16bf a0 = cat8(wpl0[ch*64], wpl0[ch*64 + 1]);
    v16bf a1 = cat8(wpl1[ch*64], wpl1[ch*64 + 1]);
    const v8bf* bp0 = (const v8bf*)(bb0 + (ch << 5));
    const v8bf* bp1 = (const v8bf*)(bb1 + (ch << 5));
    v16bf bf0 = cat8(bp0[0], bp0[2]);
    v16bf bf1 = cat8(bp1[0], bp1[2]);
    acc00 = __builtin_amdgcn_wmma_f32_16x16x32_bf16(false, a0, false, bf0, (short)0, acc00, false, false);
    acc01 = __builtin_amdgcn_wmma_f32_16x16x32_bf16(false, a0, false, bf1, (short)0, acc01, false, false);
    acc10 = __builtin_amdgcn_wmma_f32_16x16x32_bf16(false, a1, false, bf0, (short)0, acc10, false, false);
    acc11 = __builtin_amdgcn_wmma_f32_16x16x32_bf16(false, a1, false, bf1, (short)0, acc11, false, false);
  }

  // ---- epilogue: C/D layout (VGPR r, lane<16: M=r; lane>=16: M=r+8; N=lane&15)
  const int rbase = hi16 ? 8 : 0;
  long pix0 = (long)y0*W_ + x0 + nn;
  float* ob       = out    + (long)b*G_*HW_ + pix0;
  const float* ab = addsrc + (long)b*G_*HW_ + pix0;
#pragma unroll
  for (int r = 0; r < 8; ++r){
    long o0 = (long)(m0*16      + rbase + r)*HW_;
    long o1 = (long)(m0*16 + 16 + rbase + r)*HW_;
    float v00 = acc00[r], v01 = acc01[r], v10 = acc10[r], v11 = acc11[r];
    if (addflag){
      v00 += ab[o0]; v01 += ab[o0 + W_];
      v10 += ab[o1]; v11 += ab[o1 + W_];
    }
    ob[o0] = v00; ob[o0 + W_] = v01;
    ob[o1] = v10; ob[o1 + W_] = v11;
  }
}

// ---------------------------------------------------------------------------
// Pointwise ConvLSTM gate update. z: [B][256][H][W], gate order i,f,g,o.
// ---------------------------------------------------------------------------
__global__ void lstm_point(const float* __restrict__ z, const float* __restrict__ bias,
                           float* __restrict__ cst, float* __restrict__ hst,
                           float* __restrict__ seqout, int t){
  long i = (long)blockIdx.x*blockDim.x + threadIdx.x;
  const long NTOT = (long)B_*F_*HW_;
  if (i >= NTOT) return;
  int b = (int)(i / (F_*HW_));
  int r = (int)(i % (F_*HW_));
  int f = r / HW_;
  int p = r % HW_;
  long zb = (long)b*G_*HW_ + (long)f*HW_ + p;
  float zi = z[zb             ] + bias[f];
  float zf = z[zb +  64L*HW_  ] + bias[ 64 + f];
  float zg = z[zb + 128L*HW_  ] + bias[128 + f];
  float zo = z[zb + 192L*HW_  ] + bias[192 + f];
  float c  = cst[i];
  float hi = fminf(fmaxf(0.2f*zi + 0.5f, 0.0f), 1.0f);
  float hf = fminf(fmaxf(0.2f*zf + 0.5f, 0.0f), 1.0f);
  float ho = fminf(fmaxf(0.2f*zo + 0.5f, 0.0f), 1.0f);
  float sg = 1.0f/(1.0f + expf(-zg));
  float cn = hf*c + hi*sg;
  float hn = ho * (1.0f/(1.0f + expf(-cn)));
  cst[i] = cn;
  hst[i] = hn;
  seqout[((long)b*T_ + t)*(long)(F_*HW_) + (long)f*HW_ + p] = hn;
}

// ---------------------------------------------------------------------------
// BatchNorm over all axes except W (80 columns), training-mode batch stats.
// Deterministic: fixed per-thread accumulation order + fixed tree reduce.
// ---------------------------------------------------------------------------
__global__ void bn_stats(const float* __restrict__ x, const float* __restrict__ gamma,
                         const float* __restrict__ beta, float* __restrict__ scale,
                         float* __restrict__ shiftv, long rows){
  int w = blockIdx.x;
  float s = 0.f, q = 0.f;
  for (long j = threadIdx.x; j < rows; j += blockDim.x){
    float v = x[j*W_ + w];
    s += v; q += v*v;
  }
  __shared__ float ss[1024], qq[1024];
  ss[threadIdx.x] = s; qq[threadIdx.x] = q;
  __syncthreads();
  for (int o = 512; o > 0; o >>= 1){
    if (threadIdx.x < o){ ss[threadIdx.x] += ss[threadIdx.x+o]; qq[threadIdx.x] += qq[threadIdx.x+o]; }
    __syncthreads();
  }
  if (threadIdx.x == 0){
    float inv = 1.0f/(float)rows;
    float mean = ss[0]*inv;
    float var  = qq[0]*inv - mean*mean;
    float sc = gamma[w] * rsqrtf(var + 1e-3f);
    scale[w]  = sc;
    shiftv[w] = beta[w] - mean*sc;
  }
}

__global__ void bn_apply(float* __restrict__ x, const float* __restrict__ scale,
                         const float* __restrict__ shiftv, long n){
  long i = (long)blockIdx.x*blockDim.x + threadIdx.x;
  if (i < n){
    int w = (int)(i % W_);
    x[i] = x[i]*scale[w] + shiftv[w];
  }
}

// ---------------------------------------------------------------------------
extern "C" void kernel_launch(void* const* d_in, const int* in_sizes, int n_in,
                              void* d_out, int out_size, void* d_ws, size_t ws_size,
                              hipStream_t stream){
  (void)in_sizes; (void)n_in; (void)out_size; (void)ws_size;
  const float* inputs = (const float*)d_in[0];
  const float* h1_0 = (const float*)d_in[1];
  const float* c1_0 = (const float*)d_in[2];
  const float* h2_0 = (const float*)d_in[3];
  const float* c2_0 = (const float*)d_in[4];
  const float* W1 = (const float*)d_in[5];
  const float* U1 = (const float*)d_in[6];
  const float* b1 = (const float*)d_in[7];
  const float* W2 = (const float*)d_in[8];
  const float* U2 = (const float*)d_in[9];
  const float* b2 = (const float*)d_in[10];
  const float* gamma = (const float*)d_in[11];
  const float* beta  = (const float*)d_in[12];

  const long NX = (long)B_*T_*F_*HW_;   // 14,745,600
  const long NS = (long)B_*F_*HW_;      //  1,228,800
  float* X2 = (float*)d_out;
  float* H1 = X2 + NX;
  float* C1 = H1 + NS;
  float* H2 = C1 + NS;
  float* C2 = H2 + NS;

  // workspace carve-up (16B alignment preserved throughout)
  float* seq1   = (float*)d_ws;                  // layer1 output / normalized seq
  float* z      = seq1 + NX;                     // gate pre-activations [B][256][H][W]
  float* scale  = z + (long)B_*G_*HW_;
  float* shiftv = scale + 80;
  __bf16* W1p = (__bf16*)(shiftv + 80);          // 16*2*512 hw
  __bf16* U1p = W1p + 16*2*512;                  // 16*18*512 hw each
  __bf16* W2p = U1p + 16*18*512;
  __bf16* U2p = W2p + 16*18*512;

  // pack weights into WMMA A-fragment order (bf16)
  pack_weights<<<(16*2*512 + 255)/256, 256, 0, stream>>>(W1, W1p,  2,  3, 2);
  pack_weights<<<(16*18*512 + 255)/256, 256, 0, stream>>>(U1, U1p, 18, 63, 6);
  pack_weights<<<(16*18*512 + 255)/256, 256, 0, stream>>>(W2, W2p, 18, 63, 6);
  pack_weights<<<(16*18*512 + 255)/256, 256, 0, stream>>>(U2, U2p, 18, 63, 6);

  // initial states live directly in their d_out regions
  hipMemcpyAsync(H1, h1_0, NS*sizeof(float), hipMemcpyDeviceToDevice, stream);
  hipMemcpyAsync(C1, c1_0, NS*sizeof(float), hipMemcpyDeviceToDevice, stream);
  hipMemcpyAsync(H2, h2_0, NS*sizeof(float), hipMemcpyDeviceToDevice, stream);
  hipMemcpyAsync(C2, c2_0, NS*sizeof(float), hipMemcpyDeviceToDevice, stream);

  dim3 cgrid(W_/16, H_/2, B_);
  const int smem1 = 32*(64*2  + 16);   //  4,608 B (layer-1 input conv, K=64)
  const int smem2 = 32*(576*2 + 16);   // 37,376 B (K=576 convs)
  const int pwBlocks = (int)((NS + 255)/256);

  // ---- layer 1 ----
  for (int t = 0; t < T_; ++t){
    conv3x3_wmma<<<cgrid, 256, smem1, stream>>>(inputs + (long)t*4*HW_, (long)T_*4*HW_,
        W1p, z, 0, z, 64, 3, 2, /*clamp*/1);
    conv3x3_wmma<<<cgrid, 256, smem2, stream>>>(H1, (long)F_*HW_,
        U1p, z, 1, z, 576, 63, 6, /*zero*/0);
    lstm_point<<<pwBlocks, 256, 0, stream>>>(z, b1, C1, H1, seq1, t);
  }

  // ---- batchnorm (per W column, training-mode stats) ----
  bn_stats<<<W_, 1024, 0, stream>>>(seq1, gamma, beta, scale, shiftv, NX / W_);
  bn_apply<<<(int)((NX + 255)/256), 256, 0, stream>>>(seq1, scale, shiftv, NX);

  // ---- layer 2 (hidden states written straight into d_out X region) ----
  for (int t = 0; t < T_; ++t){
    conv3x3_wmma<<<cgrid, 256, smem2, stream>>>(seq1 + (long)t*F_*HW_, (long)T_*F_*HW_,
        W2p, z, 0, z, 576, 63, 6, /*clamp*/1);
    conv3x3_wmma<<<cgrid, 256, smem2, stream>>>(H2, (long)F_*HW_,
        U2p, z, 1, z, 576, 63, 6, /*zero*/0);
    lstm_point<<<pwBlocks, 256, 0, stream>>>(z, b2, C2, H2, X2, t);
  }
}
